// Attn_70076686401576
// MI455X (gfx1250) — matrix operand order repaired and checked by reading, not compiled, not run on hardware
//
#include <hip/hip_runtime.h>

typedef __attribute__((ext_vector_type(16))) __bf16 v16bf;
typedef __attribute__((ext_vector_type(8)))  float  v8f;

#define N_TOK 4096
#define D_MOD 1024
#define USE_ASYNC_COPY 1

__device__ __forceinline__ unsigned short f2bf(float f) {
  unsigned int u = __float_as_uint(f);
  u += 0x7FFFu + ((u >> 16) & 1u);   // round-to-nearest-even
  return (unsigned short)(u >> 16);
}

// ---- gfx1250 async global->LDS copy (ASYNCcnt path) -----------------------
__device__ __forceinline__ void async_g2l_b128(void* lds, const void* gptr) {
#if USE_ASYNC_COPY
  unsigned loff = (unsigned)(uintptr_t)lds;                 // LDS byte offset (addr[31:0])
  unsigned long long ga = (unsigned long long)(uintptr_t)gptr;
  asm volatile("global_load_async_to_lds_b128 %0, %1, off" :: "v"(loff), "v"(ga) : "memory");
#else
  *(uint4*)lds = *(const uint4*)gptr;
#endif
}
#if USE_ASYNC_COPY
#define ASYNC_WAIT0() asm volatile("s_wait_asynccnt 0x0" ::: "memory")
#else
#define ASYNC_WAIT0()
#endif

// ---- WMMA fragment helpers ------------------------------------------------
union FragU { v16bf v; uint4 q[2]; };

// A 16x32 bf16 frag (row-major LDS). lane<16: row=lane, K=kb+0..7,kb+16..23;
// lane>=16: row=lane-16, K=kb+8..15,kb+24..31
__device__ __forceinline__ v16bf ldsA(const unsigned short* p, int stride, int m0, int kb, int lane) {
  const unsigned short* r = p + (m0 + (lane & 15)) * stride + kb + ((lane >> 4) << 3);
  FragU u;
  u.q[0] = *(const uint4*)(r);
  u.q[1] = *(const uint4*)(r + 16);
  return u.v;
}
// B 32x16 bf16 frag from K-contiguous storage Bt[n][k].
__device__ __forceinline__ v16bf ldsB(const unsigned short* p, int stride, int n0, int kb, int lane) {
  const unsigned short* r = p + (n0 + (lane & 15)) * stride + kb + ((lane >> 4) << 3);
  FragU u;
  u.q[0] = *(const uint4*)(r);
  u.q[1] = *(const uint4*)(r + 16);
  return u.v;
}
__device__ __forceinline__ v8f wmma_bf16(v16bf a, v16bf b, v8f c) {
  return __builtin_amdgcn_wmma_f32_16x16x32_bf16(false, a, false, b, (short)0, c, false, false);
}

// ---------------------------------------------------------------------------
__global__ __launch_bounds__(256) void cvt_bf16_kernel(const float* __restrict__ s,
                                                       unsigned short* __restrict__ d, int n) {
  int i = blockIdx.x * blockDim.x + threadIdx.x;
  if (i < n) d[i] = f2bf(s[i]);
}

// f32 [R][C] -> bf16 [C][R], tiled 64x64 through LDS, coalesced both sides
__global__ __launch_bounds__(256) void cvtT_kernel(const float* __restrict__ src,
                                                   unsigned short* __restrict__ dst,
                                                   int R, int C) {
  __shared__ __align__(16) unsigned short tile[64 * 72];
  const int r0 = blockIdx.y * 64, c0 = blockIdx.x * 64;
  const int t = threadIdx.x;
  for (int i = 0; i < 4; i++) {
    int e = i * 256 + t;                  // 0..1023, chunks of 4 floats
    int r = e >> 4, c4 = (e & 15) << 2;
    float4 v = *(const float4*)&src[(size_t)(r0 + r) * C + c0 + c4];
    union { unsigned short s4[4]; uint2 q; } u;
    u.s4[0] = f2bf(v.x); u.s4[1] = f2bf(v.y); u.s4[2] = f2bf(v.z); u.s4[3] = f2bf(v.w);
    *(uint2*)&tile[r * 72 + c4] = u.q;
  }
  __syncthreads();
  for (int i = 0; i < 2; i++) {
    int e = i * 256 + t;                  // 0..511, 8-elem chunks along dst row
    int c = e >> 3, rc = (e & 7) << 3;
    union { unsigned short s8[8]; uint4 q; } u;
    for (int j = 0; j < 8; j++) u.s8[j] = tile[(rc + j) * 72 + c];
    *(uint4*)&dst[(size_t)(c0 + c) * R + r0 + rc] = u.q;
  }
}

// per-head u16 transpose: vh[h][4096][64] -> vT[h][64][4096]
__global__ __launch_bounds__(256) void transT_v_kernel(const unsigned short* __restrict__ vh,
                                                       unsigned short* __restrict__ vT) {
  __shared__ __align__(16) unsigned short tile[64 * 72];
  const int h = blockIdx.y, r0 = blockIdx.x * 64;
  const unsigned short* src = vh + (size_t)h * N_TOK * 64;
  unsigned short* dst = vT + (size_t)h * 64 * N_TOK;
  const int t = threadIdx.x;
  for (int i = 0; i < 2; i++) {
    int e = i * 256 + t;                  // 512 chunks of 8 u16
    int r = e >> 3, c8 = (e & 7) << 3;
    *(uint4*)&tile[r * 72 + c8] = *(const uint4*)&src[(size_t)(r0 + r) * 64 + c8];
  }
  __syncthreads();
  for (int i = 0; i < 2; i++) {
    int e = i * 256 + t;
    int c = e >> 3, rc = (e & 7) << 3;
    union { unsigned short s8[8]; uint4 q; } u;
    for (int j = 0; j < 8; j++) u.s8[j] = tile[(rc + j) * 72 + c];
    *(uint4*)&dst[(size_t)c * N_TOK + r0 + rc] = u.q;
  }
}

// ---------------------------------------------------------------------------
// QKV GEMM: C[4096,3072] = xb @ Wqkv + bias with QKNorm epilogue.
// wT is Wqkv transposed: [3072][1024]. Double-buffered async staging.
__global__ __launch_bounds__(256) void gemm_qkv_kernel(
    const unsigned short* __restrict__ xb, const unsigned short* __restrict__ wT,
    const float* __restrict__ bias, const float* __restrict__ gq, const float* __restrict__ gk,
    unsigned short* __restrict__ qh, unsigned short* __restrict__ kh,
    unsigned short* __restrict__ vh) {
  __shared__ __align__(16) unsigned short As[2][128 * 32];
  __shared__ __align__(16) unsigned short Bt[2][64 * 32];
  __shared__ float Cs[128 * 65];

  const int t = threadIdx.x, lane = t & 31, w = t >> 5;
  const int wm = w >> 1, wn = w & 1;            // 4(M) x 2(N) waves, 32x32 each
  const int row0 = blockIdx.y * 128;
  const int cb = blockIdx.x;                    // 0..47 == one (q|k|v, head) group
  const int c0 = cb * 64;

  auto stage = [&](int b, int kt) {
    for (int i = 0; i < 2; i++) {
      int c = t * 2 + i, r = c >> 2, col = (c & 3) << 3;
      async_g2l_b128(&As[b][r * 32 + col], &xb[(size_t)(row0 + r) * 1024 + kt + col]);
    }
    int n = t >> 2, kc = (t & 3) << 3;
    async_g2l_b128(&Bt[b][n * 32 + kc], &wT[(size_t)(c0 + n) * 1024 + kt + kc]);
  };

  v8f zero8 = {0.f, 0.f, 0.f, 0.f, 0.f, 0.f, 0.f, 0.f};
  v8f acc[2][2];
  for (int i = 0; i < 2; i++) for (int j = 0; j < 2; j++) acc[i][j] = zero8;

  stage(0, 0);
  ASYNC_WAIT0();
  __syncthreads();
  for (int kt = 0; kt < 1024; kt += 32) {
    int cur = (kt >> 5) & 1;
    if (kt + 32 < 1024) stage(cur ^ 1, kt + 32);   // DMA next tile during compute
    v16bf a0 = ldsA(As[cur], 32, wm * 32, 0, lane);
    v16bf a1 = ldsA(As[cur], 32, wm * 32 + 16, 0, lane);
    v16bf b0 = ldsB(Bt[cur], 32, wn * 32, 0, lane);
    v16bf b1 = ldsB(Bt[cur], 32, wn * 32 + 16, 0, lane);
    acc[0][0] = wmma_bf16(a0, b0, acc[0][0]);
    acc[0][1] = wmma_bf16(a0, b1, acc[0][1]);
    acc[1][0] = wmma_bf16(a1, b0, acc[1][0]);
    acc[1][1] = wmma_bf16(a1, b1, acc[1][1]);
    ASYNC_WAIT0();
    __syncthreads();
  }
  for (int mi = 0; mi < 2; mi++)
    for (int ni = 0; ni < 2; ni++)
      for (int r = 0; r < 8; r++) {
        int row = wm * 32 + mi * 16 + r + ((lane >> 4) << 3);
        int col = wn * 32 + ni * 16 + (lane & 15);
        Cs[row * 65 + col] = acc[mi][ni][r];
      }
  __syncthreads();
  if (t < 128) {
    const int which = cb >> 4, head = cb & 15;
    const int tok = row0 + t;
    const float* crow = &Cs[t * 65];
    if (which < 2) {  // Q or K: RMSNorm over this 64-wide head group
      float ss = 0.f;
      for (int c = 0; c < 64; c++) { float x = crow[c] + bias[c0 + c]; ss += x * x; }
      float sc = rsqrtf(ss * (1.f / 64.f) + 1e-6f);
      const float* g = (which == 0) ? gq : gk;
      unsigned short* dst = ((which == 0) ? qh : kh) + (size_t)head * N_TOK * 64 + (size_t)tok * 64;
      for (int c = 0; c < 64; c++) {
        float x = crow[c] + bias[c0 + c];
        dst[c] = f2bf(x * sc * g[c]);
      }
    } else {
      unsigned short* dst = vh + (size_t)head * N_TOK * 64 + (size_t)tok * 64;
      for (int c = 0; c < 64; c++) dst[c] = f2bf(crow[c] + bias[c0 + c]);
    }
  }
}

// ---------------------------------------------------------------------------
// Flash attention per (head, 64-query tile). Additive +1.0 frame mask (all
// 4096 keys participate). 4 waves 2x2 of 32x32; double-buffered async K/V.
__global__ __launch_bounds__(128) void attn_kernel(
    const unsigned short* __restrict__ qh, const unsigned short* __restrict__ kh,
    const unsigned short* __restrict__ vT, unsigned short* __restrict__ obf,
    const int* __restrict__ tpf_p) {
  __shared__ __align__(16) char smem[61696];
  // runtime-computed LDS pointers (no constant-initialized pointer arrays!)
  auto KsP = [&](int b) { return (unsigned short*)(smem + b * 8192); };
  auto VtP = [&](int b) { return (unsigned short*)(smem + 16384 + b * 9216); };
  unsigned short* Qs   = (unsigned short*)(smem + 8192);   // aliases K buf 1
  float* Ss            = (float*)(smem + 34816);           // 64 x 65
  unsigned short* Ps   = (unsigned short*)(smem + 51456);  // 64 x 64 bf16
  float* alphaS        = (float*)(smem + 59648);
  float* mrow          = (float*)(smem + 59904);
  float* lrow          = (float*)(smem + 60160);
  float* pmax          = (float*)(smem + 60416);           // [64][2]
  float* psum          = (float*)(smem + 60928);           // [64][2]
  float* linvS         = (float*)(smem + 61440);

  const int t = threadIdx.x, lane = t & 31, w = t >> 5;
  const int wm = w >> 1, wn = w & 1;
  const int h = blockIdx.y, q0 = blockIdx.x * 64;
  const int tpf = tpf_p[0];
  const size_t hb = (size_t)h * N_TOK * 64;
  const int qf_lo = q0 / tpf, qf_hi = (q0 + 63) / tpf;

  auto stageKV = [&](int b, int k0) {
    unsigned short* Kb = KsP(b);
    unsigned short* Vb = VtP(b);
    for (int i = 0; i < 4; i++) {
      int c = i * 128 + t, r = c >> 3, col = (c & 7) << 3;
      async_g2l_b128(&Kb[r * 64 + col], &kh[hb + (size_t)(k0 + r) * 64 + col]);
    }
    for (int i = 0; i < 4; i++) {
      int c = i * 128 + t, d = c >> 3, kc = (c & 7) << 3;
      async_g2l_b128(&Vb[d * 72 + kc], &vT[hb + (size_t)d * N_TOK + k0 + kc]);
    }
  };

  // stage Q and first K/V tile together
  for (int i = 0; i < 4; i++) {
    int c = i * 128 + t, r = c >> 3, col = (c & 7) << 3;
    async_g2l_b128(&Qs[r * 64 + col], &qh[hb + (size_t)(q0 + r) * 64 + col]);
  }
  stageKV(0, 0);
  ASYNC_WAIT0();
  __syncthreads();

  v16bf qa[2][2];
  for (int mi = 0; mi < 2; mi++)
    for (int ks = 0; ks < 2; ks++)
      qa[mi][ks] = ldsA(Qs, 64, wm * 32 + mi * 16, ks * 32, lane);
  if (t < 64) { mrow[t] = -INFINITY; lrow[t] = 0.f; }
  __syncthreads();   // Q frags in regs; Qs region now reusable as K buf 1

  v8f zero8 = {0.f, 0.f, 0.f, 0.f, 0.f, 0.f, 0.f, 0.f};
  v8f oacc[2][2];
  for (int i = 0; i < 2; i++) for (int j = 0; j < 2; j++) oacc[i][j] = zero8;

  const int sr = t & 63, half = t >> 6;

  for (int kt = 0; kt < 64; kt++) {
    const int k0 = kt * 64, cur = kt & 1;
    if (kt + 1 < 64) stageKV(cur ^ 1, k0 + 64);      // DMA next tile during compute

    // S = Q @ K^T
    unsigned short* Kb = KsP(cur);
    v8f sacc[2][2];
    for (int i = 0; i < 2; i++) for (int j = 0; j < 2; j++) sacc[i][j] = zero8;
    for (int ks = 0; ks < 2; ks++) {
      v16bf b0 = ldsB(Kb, 64, wn * 32, ks * 32, lane);
      v16bf b1 = ldsB(Kb, 64, wn * 32 + 16, ks * 32, lane);
      sacc[0][0] = wmma_bf16(qa[0][ks], b0, sacc[0][0]);
      sacc[0][1] = wmma_bf16(qa[0][ks], b1, sacc[0][1]);
      sacc[1][0] = wmma_bf16(qa[1][ks], b0, sacc[1][0]);
      sacc[1][1] = wmma_bf16(qa[1][ks], b1, sacc[1][1]);
    }
    // scale + additive frame mask (uniform per tile when 64 | tokens_per_frame)
    const int kf_lo = k0 / tpf, kf_hi = (k0 + 63) / tpf;
    const bool uni = (qf_lo == qf_hi) && (kf_lo == kf_hi);
    const float madd = (kf_lo > qf_lo) ? 1.f : 0.f;
    for (int mi = 0; mi < 2; mi++)
      for (int ni = 0; ni < 2; ni++)
        for (int r = 0; r < 8; r++) {
          int row = wm * 32 + mi * 16 + r + ((lane >> 4) << 3);
          int col = wn * 32 + ni * 16 + (lane & 15);
          float s = sacc[mi][ni][r] * 0.125f;
          if (uni) s += madd;
          else if ((k0 + col) / tpf > (q0 + row) / tpf) s += 1.0f;
          Ss[row * 65 + col] = s;
        }
    __syncthreads();

    // online softmax, 2 threads per row
    const float* srow = &Ss[sr * 65 + half * 32];
    float tm = srow[0];
    for (int c = 1; c < 32; c++) tm = fmaxf(tm, srow[c]);
    pmax[sr * 2 + half] = tm;
    __syncthreads();
    if (half == 0) {
      float tmax = fmaxf(pmax[2 * sr], pmax[2 * sr + 1]);
      float mold = mrow[sr];
      float mnew = fmaxf(mold, tmax);
      alphaS[sr] = __expf(mold - mnew);
      mrow[sr] = mnew;
    }
    __syncthreads();
    {
      float mnew = mrow[sr];
      float ps = 0.f;
      unsigned short* prow = &Ps[sr * 64 + half * 32];
      for (int c = 0; c < 32; c++) {
        float p = __expf(srow[c] - mnew);
        prow[c] = f2bf(p);
        ps += p;
      }
      psum[sr * 2 + half] = ps;
    }
    __syncthreads();
    if (half == 0) lrow[sr] = lrow[sr] * alphaS[sr] + psum[2 * sr] + psum[2 * sr + 1];

    // rescale running O, then O += P @ V
    for (int mi = 0; mi < 2; mi++)
      for (int r = 0; r < 8; r++) {
        float a = alphaS[wm * 32 + mi * 16 + r + ((lane >> 4) << 3)];
        oacc[mi][0][r] *= a;
        oacc[mi][1][r] *= a;
      }
    unsigned short* Vb = VtP(cur);
    for (int ks = 0; ks < 2; ks++) {
      v16bf pa0 = ldsA(Ps, 64, wm * 32, ks * 32, lane);
      v16bf pa1 = ldsA(Ps, 64, wm * 32 + 16, ks * 32, lane);
      v16bf vb0 = ldsB(Vb, 72, wn * 32, ks * 32, lane);
      v16bf vb1 = ldsB(Vb, 72, wn * 32 + 16, ks * 32, lane);
      oacc[0][0] = wmma_bf16(pa0, vb0, oacc[0][0]);
      oacc[0][1] = wmma_bf16(pa0, vb1, oacc[0][1]);
      oacc[1][0] = wmma_bf16(pa1, vb0, oacc[1][0]);
      oacc[1][1] = wmma_bf16(pa1, vb1, oacc[1][1]);
    }
    ASYNC_WAIT0();
    __syncthreads();
  }
  if (t < 64) linvS[t] = 1.f / lrow[t];
  __syncthreads();
  for (int mi = 0; mi < 2; mi++)
    for (int ni = 0; ni < 2; ni++)
      for (int r = 0; r < 8; r++) {
        int row = wm * 32 + mi * 16 + r + ((lane >> 4) << 3);
        int col = wn * 32 + ni * 16 + (lane & 15);
        float v = oacc[mi][ni][r] * linvS[row];
        obf[(size_t)(q0 + row) * 1024 + h * 64 + col] = f2bf(v);
      }
}

// ---------------------------------------------------------------------------
// Output GEMM: out[4096,1024] = obf @ Wout + bout. wT = Wout^T [1024][1024].
__global__ __launch_bounds__(256) void gemm_out_kernel(
    const unsigned short* __restrict__ ab, const unsigned short* __restrict__ wT,
    const float* __restrict__ bias, float* __restrict__ out) {
  __shared__ __align__(16) unsigned short As[2][128 * 32];
  __shared__ __align__(16) unsigned short Bt[2][64 * 32];

  const int t = threadIdx.x, lane = t & 31, w = t >> 5;
  const int wm = w >> 1, wn = w & 1;
  const int row0 = blockIdx.y * 128;
  const int c0 = blockIdx.x * 64;

  auto stage = [&](int b, int kt) {
    for (int i = 0; i < 2; i++) {
      int c = t * 2 + i, r = c >> 2, col = (c & 3) << 3;
      async_g2l_b128(&As[b][r * 32 + col], &ab[(size_t)(row0 + r) * 1024 + kt + col]);
    }
    int n = t >> 2, kc = (t & 3) << 3;
    async_g2l_b128(&Bt[b][n * 32 + kc], &wT[(size_t)(c0 + n) * 1024 + kt + kc]);
  };

  v8f zero8 = {0.f, 0.f, 0.f, 0.f, 0.f, 0.f, 0.f, 0.f};
  v8f acc[2][2];
  for (int i = 0; i < 2; i++) for (int j = 0; j < 2; j++) acc[i][j] = zero8;

  stage(0, 0);
  ASYNC_WAIT0();
  __syncthreads();
  for (int kt = 0; kt < 1024; kt += 32) {
    int cur = (kt >> 5) & 1;
    if (kt + 32 < 1024) stage(cur ^ 1, kt + 32);
    v16bf a0 = ldsA(As[cur], 32, wm * 32, 0, lane);
    v16bf a1 = ldsA(As[cur], 32, wm * 32 + 16, 0, lane);
    v16bf b0 = ldsB(Bt[cur], 32, wn * 32, 0, lane);
    v16bf b1 = ldsB(Bt[cur], 32, wn * 32 + 16, 0, lane);
    acc[0][0] = wmma_bf16(a0, b0, acc[0][0]);
    acc[0][1] = wmma_bf16(a0, b1, acc[0][1]);
    acc[1][0] = wmma_bf16(a1, b0, acc[1][0]);
    acc[1][1] = wmma_bf16(a1, b1, acc[1][1]);
    ASYNC_WAIT0();
    __syncthreads();
  }
  for (int mi = 0; mi < 2; mi++)
    for (int ni = 0; ni < 2; ni++)
      for (int r = 0; r < 8; r++) {
        int row = row0 + wm * 32 + mi * 16 + r + ((lane >> 4) << 3);
        int col = c0 + wn * 32 + ni * 16 + (lane & 15);
        out[(size_t)row * 1024 + col] = acc[mi][ni][r] + bias[col];
      }
}

// ---------------------------------------------------------------------------
extern "C" void kernel_launch(void* const* d_in, const int* in_sizes, int n_in,
                              void* d_out, int out_size, void* d_ws, size_t ws_size,
                              hipStream_t stream) {
  (void)in_sizes; (void)n_in; (void)out_size; (void)ws_size;
  const float* x    = (const float*)d_in[0];
  const float* Wqkv = (const float*)d_in[1];
  const float* bqkv = (const float*)d_in[2];
  const float* Wout = (const float*)d_in[3];
  const float* bout = (const float*)d_in[4];
  const float* gq   = (const float*)d_in[5];
  const float* gk   = (const float*)d_in[6];
  const int*   tpf  = (const int*)d_in[7];
  float* outp = (float*)d_out;

  char* ws = (char*)d_ws;
  unsigned short* xb     = (unsigned short*)(ws);                         // 8 MB
  unsigned short* wqkvT  = (unsigned short*)(ws + (size_t)(8u  << 20));   // 6 MB  [3072][1024]
  unsigned short* woutT  = (unsigned short*)(ws + (size_t)(14u << 20));   // 2 MB  [1024][1024]
  unsigned short* qh     = (unsigned short*)(ws + (size_t)(16u << 20));   // 8 MB
  unsigned short* kh     = (unsigned short*)(ws + (size_t)(24u << 20));   // 8 MB
  unsigned short* vh     = (unsigned short*)(ws + (size_t)(32u << 20));   // 8 MB
  unsigned short* vT     = (unsigned short*)(ws + (size_t)(40u << 20));   // 8 MB  [h][64][4096]
  unsigned short* obf    = (unsigned short*)(ws + (size_t)(48u << 20));   // 8 MB

  cvt_bf16_kernel<<<(4096 * 1024) / 256, 256, 0, stream>>>(x, xb, 4096 * 1024);
  cvtT_kernel<<<dim3(48, 16), 256, 0, stream>>>(Wqkv, wqkvT, 1024, 3072);
  cvtT_kernel<<<dim3(16, 16), 256, 0, stream>>>(Wout, woutT, 1024, 1024);

  gemm_qkv_kernel<<<dim3(48, 32), 256, 0, stream>>>(xb, wqkvT, bqkv, gq, gk, qh, kh, vh);
  transT_v_kernel<<<dim3(64, 16), 256, 0, stream>>>(vh, vT);
  attn_kernel<<<dim3(64, 16), 128, 0, stream>>>(qh, kh, vT, obf, tpf);
  gemm_out_kernel<<<dim3(16, 32), 256, 0, stream>>>(obf, woutT, bout, outp);
}
